// TSPHead_10926396801617
// MI455X (gfx1250) — matrix operand order repaired and checked by reading, not compiled, not run on hardware
//
#include <hip/hip_runtime.h>
#include <hip/hip_bf16.h>
#include <math.h>

// ---------------------------------------------------------------------------
// TSPHead for MI455X (gfx1250, wave32, WMMA).
//
// Pipeline (all on `stream`):
//   1) convert_w      : W_fuse (256x128 f32) -> wT (f16, transposed [j][k]) in ws
//   2) zero_stats     : zero 256-float stats accumulators in ws
//   3) gemm_stats     : WMMA f16 GEMM (f32 accum), per-column sum / sum-sq.
//                       Weights staged in LDS (padded stride) once per block.
//   4) finalize_stats : mu, rstd = rsqrt(var+eps)
//   5) gemm_heads     : WMMA GEMM again, BN+ReLU, bbox/cls heads -> preds
//   6) kth_select     : exact 33rd-smallest masked d2 per box (bitwise bsearch)
//   7) assign         : per-point gated argmin -> assigned (-1 or box index)
//
// b_fuse is skipped: a constant column shift cancels exactly under BatchNorm.
// ---------------------------------------------------------------------------

#define FLOAT_MAX_C 1.0e8f
#define BN_EPS_C 1e-5f

// LDS weight tile: 128 rows (output cols j) x 256 K, padded to 264 halfs/row.
// 264*2B = 132 dwords; 132 % 64 = 4 -> B-fragment lanes land on distinct banks.
#define WROW 264
#define WLDS_HALFS (128 * WROW)
#define WLDS_BYTES (WLDS_HALFS * 2)

typedef __attribute__((ext_vector_type(16))) _Float16 v16h;
typedef __attribute__((ext_vector_type(8)))  _Float16 v8h;
typedef __attribute__((ext_vector_type(8)))  float    v8f;

__device__ __forceinline__ void load8(const float* __restrict__ p, float f[8]) {
    const float4 a = *reinterpret_cast<const float4*>(p);
    const float4 b = *reinterpret_cast<const float4*>(p + 4);
    f[0] = a.x; f[1] = a.y; f[2] = a.z; f[3] = a.w;
    f[4] = b.x; f[5] = b.y; f[6] = b.z; f[7] = b.w;
}

// Stage the 64KB f16 transposed weight matrix from global (ws) into LDS,
// restriding 256 -> WROW halfs per row. 128 threads x 32 iters of 16B copies.
__device__ __forceinline__ void stage_weights(const _Float16* __restrict__ wT,
                                              _Float16* wlds, int tid) {
    for (int idx = tid; idx < 128 * 32; idx += 128) {
        const int row = idx >> 5, c = idx & 31;          // 32 x 16B chunks per row
        const v8h v = *reinterpret_cast<const v8h*>(wT + row * 256 + c * 8);
        *reinterpret_cast<v8h*>(wlds + row * WROW + c * 8) = v;
    }
}

// One wave computes a 16-row x 128-col tile of fused = combined @ W_fuse.
// A layout (f16 16x32): lanes 0-15 -> M=lane, K 0-15; lanes 16-31 -> M=lane-16,
//   K 16-31; 16 contiguous K per lane (2 per VGPR), matching B.
// B layout (f16 32x16): N = lane&15; lanes 0-15 hold K 0-15, lanes 16-31 hold
//   K 16-31, 2 K per VGPR -> 16 contiguous f16 read from the LDS weight tile.
__device__ __forceinline__ void gemm_tile(const float* __restrict__ bb,
                                          const float* __restrict__ tx,
                                          const int*   __restrict__ bidx,
                                          const _Float16* wlds,
                                          int tileBase, int N, v8f* acc) {
    const int lane = threadIdx.x & 31;
    const int mloc = lane & 15;
    const int ko   = (lane < 16) ? 0 : 16;   // A-fragment K sub-offset
    const int ksB  = (lane < 16) ? 0 : 16;   // B-fragment K sub-offset
    int m = tileBase + mloc;
    if (m >= N) m = N - 1;                   // clamp; stats/stores mask later
    const int bi = bidx[m];
    const float* brow = bb + (size_t)m * 128;
    const float* trow = tx + (size_t)bi * 128;

#pragma unroll
    for (int kb = 0; kb < 256; kb += 32) {
        const float* src = (kb < 128) ? (brow + kb) : (trow + (kb - 128));
        float f0[8], f1[8];
        load8(src + ko, f0);
        load8(src + ko + 8, f1);
        v16h a;
#pragma unroll
        for (int i = 0; i < 8; ++i) {
            a[i]     = (_Float16)f0[i];
            a[8 + i] = (_Float16)f1[i];
        }
#pragma unroll
        for (int nt = 0; nt < 8; ++nt) {
            const _Float16* wp = wlds + (nt * 16 + mloc) * WROW + kb + ksB;
            const v8h b0 = *reinterpret_cast<const v8h*>(wp);      // ds_load_b128
            const v8h b1 = *reinterpret_cast<const v8h*>(wp + 8);  // ds_load_b128
            v16h b;
#pragma unroll
            for (int i = 0; i < 8; ++i) { b[i] = b0[i]; b[8 + i] = b1[i]; }
            acc[nt] = __builtin_amdgcn_wmma_f32_16x16x32_f16(
                false, a, false, b, (short)0, acc[nt], false, false);
        }
    }
}

__global__ void __launch_bounds__(256) convert_w_kernel(const float* __restrict__ W,
                                                        _Float16* __restrict__ wT) {
    const int idx = blockIdx.x * 256 + threadIdx.x;   // 0..32767
    const int j = idx >> 8, k = idx & 255;
    wT[idx] = (_Float16)W[k * 128 + j];               // wT[j*256 + k]
}

__global__ void zero_stats_kernel(float* __restrict__ p) {
    p[threadIdx.x] = 0.f;                              // 256 floats: sum+sumsq
}

__global__ void __launch_bounds__(128) gemm_stats_kernel(
    const float* __restrict__ bb, const float* __restrict__ tx,
    const int* __restrict__ bidx, const _Float16* __restrict__ wT,
    float* __restrict__ gsum, float* __restrict__ gsq, int N) {
    extern __shared__ char smem[];
    _Float16* wlds = (_Float16*)smem;                  // WLDS_BYTES
    float* ls = (float*)(smem + WLDS_BYTES);           // 128 floats
    float* lq = ls + 128;                              // 128 floats
    const int tid = threadIdx.x;
    ls[tid] = 0.f; lq[tid] = 0.f;
    stage_weights(wT, wlds, tid);
    __syncthreads();

    const int wave = tid >> 5, lane = tid & 31;
    const int tileBase = (blockIdx.x * 4 + wave) * 16;
    if (tileBase < N) {
        v8f acc[8] = {};
        gemm_tile(bb, tx, bidx, wlds, tileBase, N, acc);
        const int rhi = (lane < 16) ? 0 : 8;  // C/D layout: lanes>=16 hold M+8
#pragma unroll
        for (int nt = 0; nt < 8; ++nt) {
            float s = 0.f, q = 0.f;
#pragma unroll
            for (int r = 0; r < 8; ++r) {
                const int row = tileBase + rhi + r;
                const float v = (row < N) ? acc[nt][r] : 0.f;
                s += v; q += v * v;
            }
            s += __shfl_xor(s, 16, 32);       // combine lanes l and l^16 (same col)
            q += __shfl_xor(q, 16, 32);
            if (lane < 16) {
                const int col = nt * 16 + lane;
                atomicAdd(&ls[col], s);
                atomicAdd(&lq[col], q);
            }
        }
    }
    __syncthreads();
    atomicAdd(&gsum[tid], ls[tid]);
    atomicAdd(&gsq[tid],  lq[tid]);
}

__global__ void finalize_stats_kernel(const float* __restrict__ gsum,
                                      const float* __restrict__ gsq,
                                      float* __restrict__ mu,
                                      float* __restrict__ rstd, int N) {
    const int j = threadIdx.x;
    const float m = gsum[j] / (float)N;
    const float v = gsq[j] / (float)N - m * m;
    mu[j] = m;
    rstd[j] = rsqrtf(v + BN_EPS_C);
}

__global__ void __launch_bounds__(128) gemm_heads_kernel(
    const float* __restrict__ bb, const float* __restrict__ tx,
    const int* __restrict__ bidx, const _Float16* __restrict__ wT,
    const float* __restrict__ mu, const float* __restrict__ rstd,
    const float* __restrict__ gamma, const float* __restrict__ beta,
    const float* __restrict__ Wb, const float* __restrict__ bB,
    const float* __restrict__ Wc, const float* __restrict__ bC,
    float* __restrict__ preds, int N) {
    extern __shared__ char smem[];
    _Float16* wlds = (_Float16*)smem;                       // WLDS_BYTES
    float* xs = (float*)(smem + WLDS_BYTES);                // [4][16][132]
    float* wh = xs + 4 * 16 * 132;                          // [128][9]
    float* bh = wh + 128 * 9;                               // [9]
    const int tid = threadIdx.x;
    stage_weights(wT, wlds, tid);
    for (int idx = tid; idx < 128 * 9; idx += 128) {
        const int j = idx / 9, t = idx % 9;
        wh[j * 9 + t] = (t < 7) ? Wb[j * 7 + t] : Wc[j * 2 + (t - 7)];
    }
    if (tid < 9) bh[tid] = (tid < 7) ? bB[tid] : bC[tid - 7];
    __syncthreads();

    const int wave = tid >> 5, lane = tid & 31;
    const int tileBase = (blockIdx.x * 4 + wave) * 16;
    const bool active = (tileBase < N);
    if (active) {
        v8f acc[8] = {};
        gemm_tile(bb, tx, bidx, wlds, tileBase, N, acc);
        const int rhi = (lane < 16) ? 0 : 8;
#pragma unroll
        for (int nt = 0; nt < 8; ++nt) {
            const int col = nt * 16 + (lane & 15);
            const float m_ = mu[col], rs = rstd[col];
            const float gg = gamma[col], be = beta[col];
#pragma unroll
            for (int r = 0; r < 8; ++r) {
                const float xn = fmaxf(0.f, (acc[nt][r] - m_) * rs * gg + be);
                xs[(wave * 16 + rhi + r) * 132 + col] = xn;
            }
        }
    }
    __syncthreads();
    if (active) {
        for (int task = lane; task < 16 * 9; task += 32) {
            const int row = task / 9, o = task % 9;
            const int grow = tileBase + row;
            if (grow < N) {
                float s = bh[o];
#pragma unroll 4
                for (int j = 0; j < 128; ++j)
                    s += xs[(wave * 16 + row) * 132 + j] * wh[j * 9 + o];
                if (o >= 3 && o < 6) s = expf(s);   // size dims go through exp
                preds[(size_t)grow * 9 + o] = s;
            }
        }
    }
}

// Exact k-th smallest (k = 33) masked d2 per box via binary search on the
// float bit pattern (monotonic for non-negative floats). Non-matching points
// contribute FLOAT_MAX; if fewer than k points match, the search converges to
// bits(FLOAT_MAX), reproducing the reference's fill value exactly.
__global__ void __launch_bounds__(256) kth_select_kernel(
    const float* __restrict__ gt_boxes, const int* __restrict__ gt_labels,
    const float* __restrict__ pl0, const float* __restrict__ pl1,
    int nl0, int nl1, float* __restrict__ kth_out) {
    __shared__ unsigned s_lo, s_hi;
    __shared__ int red[256];
    const int b = blockIdx.x, tid = threadIdx.x;
    const float cx = gt_boxes[b * 7 + 0];
    const float cy = gt_boxes[b * 7 + 1];
    const float cz = gt_boxes[b * 7 + 2];
    const int req = (gt_labels[b] != 0) ? 1 : 0;   // LABEL2LEVEL = (0, 1)
    const float* P = req ? pl1 : pl0;
    const int M = req ? nl1 : nl0;
    const int Ntot = nl0 + nl1;
    const int K = (33 < Ntot) ? 33 : Ntot;          // TOP_PTS_THRESHOLD + 1
    if (tid == 0) { s_lo = 0u; s_hi = __float_as_uint(FLOAT_MAX_C); }
    __syncthreads();
    for (;;) {
        const unsigned lo = s_lo, hi = s_hi;
        if (lo >= hi) break;
        const unsigned mid = (lo + hi) >> 1;
        const float fmid = __uint_as_float(mid);
        int c = 0;
        for (int i = tid; i < M; i += 256) {
            const float dx = P[i * 3 + 0] - cx;
            const float dy = P[i * 3 + 1] - cy;
            const float dz = P[i * 3 + 2] - cz;
            const float d2 = dx * dx + dy * dy + dz * dz;
            c += (d2 <= fmid) ? 1 : 0;
        }
        red[tid] = c;
        __syncthreads();
        for (int off = 128; off > 0; off >>= 1) {
            if (tid < off) red[tid] += red[tid + off];
            __syncthreads();
        }
        if (tid == 0) { if (red[0] >= K) s_hi = mid; else s_lo = mid + 1; }
        __syncthreads();
    }
    if (tid == 0) kth_out[b] = __uint_as_float(s_lo);
}

__global__ void __launch_bounds__(256) assign_kernel(
    const float* __restrict__ gt_boxes, const int* __restrict__ gt_labels,
    const float* __restrict__ kth,
    const float* __restrict__ pl0, const float* __restrict__ pl1,
    int nl0, int nl1, int nb, float* __restrict__ out) {
    __shared__ float sc[128][3];
    __shared__ float sk[128];
    __shared__ int   sr[128];
    const int tid = threadIdx.x;
    if (tid < nb) {
        sc[tid][0] = gt_boxes[tid * 7 + 0];
        sc[tid][1] = gt_boxes[tid * 7 + 1];
        sc[tid][2] = gt_boxes[tid * 7 + 2];
        sk[tid] = kth[tid];
        sr[tid] = (gt_labels[tid] != 0) ? 1 : 0;   // LABEL2LEVEL
    }
    __syncthreads();
    const int i = blockIdx.x * 256 + tid;
    const int N = nl0 + nl1;
    if (i >= N) return;
    int lvl; float px, py, pz;
    if (i < nl0) {
        lvl = 0; px = pl0[i*3]; py = pl0[i*3+1]; pz = pl0[i*3+2];
    } else {
        const int j = i - nl0;
        lvl = 1; px = pl1[j*3]; py = pl1[j*3+1]; pz = pl1[j*3+2];
    }
    float bestAll = 3.4e38f; int bestAllIdx = 0;   // argmin over raw d2
    float bestG = FLOAT_MAX_C; int bestGIdx = 0;   // argmin over gated d2
    for (int b = 0; b < nb; ++b) {
        const float dx = px - sc[b][0];
        const float dy = py - sc[b][1];
        const float dz = pz - sc[b][2];
        const float d2 = dx * dx + dy * dy + dz * dz;
        if (d2 < bestAll) { bestAll = d2; bestAllIdx = b; }   // strict <: first min
        const float gv = (lvl == sr[b] && d2 < sk[b]) ? d2 : FLOAT_MAX_C;
        if (gv < bestG) { bestG = gv; bestGIdx = b; }
    }
    const int res = (bestG < FLOAT_MAX_C && bestGIdx == bestAllIdx) ? bestGIdx : -1;
    out[i] = (float)res;
}

extern "C" void kernel_launch(void* const* d_in, const int* in_sizes, int n_in,
                              void* d_out, int out_size, void* d_ws, size_t ws_size,
                              hipStream_t stream) {
    const float* backbone = (const float*)d_in[0];
    const float* text     = (const float*)d_in[1];
    const float* Wfuse    = (const float*)d_in[2];
    // d_in[3] = b_fuse: unused (constant column shift cancels under BatchNorm)
    const float* gamma    = (const float*)d_in[4];
    const float* beta     = (const float*)d_in[5];
    const float* Wb       = (const float*)d_in[6];
    const float* bB       = (const float*)d_in[7];
    const float* Wc       = (const float*)d_in[8];
    const float* bC       = (const float*)d_in[9];
    const float* pl0      = (const float*)d_in[10];
    const float* pl1      = (const float*)d_in[11];
    const int*   bidx     = (const int*)d_in[12];
    const int*   glab     = (const int*)d_in[13];
    const float* gbox     = (const float*)d_in[14];

    const int N   = in_sizes[0] / 128;
    const int nl0 = in_sizes[10] / 3;
    const int nl1 = in_sizes[11] / 3;
    const int nb  = in_sizes[14] / 7;   // 128 boxes

    char* ws = (char*)d_ws;
    _Float16* wT  = (_Float16*)ws;                   // 64 KB transposed f16 weights
    float* gsum   = (float*)(ws + 65536);            // 128 floats
    float* gsq    = (float*)(ws + 65536 + 512);      // 128 floats (contiguous w/ gsum)
    float* mu     = (float*)(ws + 65536 + 1024);
    float* rstd   = (float*)(ws + 65536 + 1536);
    float* kth    = (float*)(ws + 65536 + 2048);

    float* preds    = (float*)d_out;
    float* assigned = preds + (size_t)N * 9;

    convert_w_kernel<<<128, 256, 0, stream>>>(Wfuse, wT);
    zero_stats_kernel<<<1, 256, 0, stream>>>(gsum);

    const int tiles  = (N + 15) / 16;
    const int blocks = (tiles + 3) / 4;              // 4 waves (tiles) per block
    const size_t shm_stats = WLDS_BYTES + 256 * sizeof(float);
    const size_t shm_heads = WLDS_BYTES + (4 * 16 * 132 + 128 * 9 + 9) * sizeof(float);
    gemm_stats_kernel<<<blocks, 128, shm_stats, stream>>>(backbone, text, bidx, wT,
                                                          gsum, gsq, N);
    finalize_stats_kernel<<<1, 128, 0, stream>>>(gsum, gsq, mu, rstd, N);
    gemm_heads_kernel<<<blocks, 128, shm_heads, stream>>>(backbone, text, bidx, wT,
                                                          mu, rstd, gamma, beta,
                                                          Wb, bB, Wc, bC, preds, N);
    kth_select_kernel<<<nb, 256, 0, stream>>>(gbox, glab, pl0, pl1, nl0, nl1, kth);
    assign_kernel<<<(N + 255) / 256, 256, 0, stream>>>(gbox, glab, kth, pl0, pl1,
                                                       nl0, nl1, nb, assigned);
}
